// Match_30674656428431
// MI455X (gfx1250) — matrix operand order repaired and checked by reading, not compiled, not run on hardware
//
#include <hip/hip_runtime.h>
#include <math.h>

// ---------------------------------------------------------------------------
// BiMPM-like model for MI455X (gfx1250, wave32, WMMA).
// Shapes: B=64, ST=512, SL=64, D=256, H=256, L=1, NC=20.
// Strategy: persistent-workgroup LSTM scans (1 WG of 32 waves per scan
// direction), bf16 WMMA 16x16x32 GEMMs with fp32 accumulate, cell state in
// registers, h in LDS, weights streamed from L2 each timestep.
// ---------------------------------------------------------------------------

typedef __bf16 bf16_t;
typedef __attribute__((ext_vector_type(16))) __bf16 v16bf;
typedef __attribute__((ext_vector_type(8)))  __bf16 v8bf;
typedef __attribute__((ext_vector_type(8)))  float  v8f;

#define HSTR 272   // LDS row stride for h (256 + 16 pad, bf16 elements)

__device__ inline float sigf(float x) { return 1.0f / (1.0f + __expf(-x)); }

// ---- WMMA fragment loaders (layouts per CDNA5 ISA 7.12.2) ------------------
// A: 16x32 bf16 (MxK). lane m=lane&15 is row; hi=lane>>4.
// elements 0..7  <- K = 8*hi .. 8*hi+7
// elements 8..15 <- K = 16+8*hi .. 16+8*hi+7
__device__ inline v16bf load_a_frag(const bf16_t* row_base, int stride,
                                    int ktile, int lane) {
  const int m = lane & 15, hi = lane >> 4;
  const bf16_t* p = row_base + (size_t)m * stride + ktile * 32;
  union { v16bf v; v8bf h[2]; } u;
  u.h[0] = *(const v8bf*)(p + 8 * hi);
  u.h[1] = *(const v8bf*)(p + 16 + 8 * hi);
  return u.v;
}

// B: 32x16 bf16 (KxN). lane n=lane&15 is column; hi selects K half.
// elements 0..7 <- K = 8*hi..8*hi+7, elements 8..15 <- K = 16+8*hi..16+8*hi+7 of weight row n.
// W stored row-major [n][k] with ld = 256 (PyTorch (4H, K) layout).
__device__ inline v16bf load_b_frag(const bf16_t* W, int ntile, int ktile,
                                    int lane) {
  const int n = lane & 15, hi = lane >> 4;
  const bf16_t* p = W + (size_t)(ntile * 16 + n) * 256 + ktile * 32;
  union { v16bf v; v8bf h[2]; } u;
  u.h[0] = *(const v8bf*)(p + 8 * hi);
  u.h[1] = *(const v8bf*)(p + 16 + 8 * hi);
  return u.v;
}

// ---- small prep kernels ----------------------------------------------------
__global__ void k_cvt_bf16(const float* __restrict__ s, bf16_t* __restrict__ d,
                           int n) {
  int i = blockIdx.x * blockDim.x + threadIdx.x;
  if (i < n) d[i] = (bf16_t)s[i];
}

__global__ void k_bias_add(const float* __restrict__ a,
                           const float* __restrict__ b,
                           float* __restrict__ d, int n) {
  int i = blockIdx.x * blockDim.x + threadIdx.x;
  if (i < n) d[i] = a[i] + b[i];
}

// out[row][d] = (bf16) emb[tok[row]][d]; n = rows*256
__global__ void k_gather(const int* __restrict__ tok,
                         const float* __restrict__ emb,
                         bf16_t* __restrict__ X, int n) {
  int i = blockIdx.x * blockDim.x + threadIdx.x;
  if (i < n) {
    int row = i >> 8, dd = i & 255;
    X[i] = (bf16_t)emb[(size_t)tok[row] * 256 + dd];
  }
}

// ---- persistent LSTM scan kernel ------------------------------------------
struct ScanDesc {
  const bf16_t* X;     // bf16 input [B][T][256] (ctx) or nullptr
  const float*  Xf;    // fp32 input [B][T][2]   (agg) or nullptr
  const bf16_t* Wih;   // bf16 (1024,256) or nullptr
  const float*  WihF;  // fp32 (1024,2)   or nullptr
  const bf16_t* Whh;   // bf16 (1024,256)
  const float*  bias;  // fp32 (1024) = bih + bhh
  bf16_t* hs;          // bf16 [B][T][256] out, or nullptr
  float*  hT;          // fp32 [B][256] out, or nullptr
  int T;
  int rev;             // 0 = forward scan, 1 = backward scan
};
struct ScanQuad { ScanDesc s[4]; };

__global__ __launch_bounds__(1024) void lstm_scan_kernel(ScanQuad q) {
  const ScanDesc d = q.s[blockIdx.x];
  __shared__ bf16_t hbuf[64 * HSTR];   // h_{t-1}, bf16, padded rows

  const int tid  = threadIdx.x;
  const int wave = tid >> 5, lane = tid & 31;
  const int mb   = wave >> 3;          // batch row block (16 rows), 0..3
  const int qd   = wave & 7;           // hidden tile-column group, 0..7
  const int hi   = lane >> 4, nloc = lane & 15;
  const int T    = d.T;

  for (int i = tid; i < 64 * HSTR; i += 1024) hbuf[i] = (bf16_t)0.0f;

  float creg[2][8];   // cell state, owned exclusively by this lane
  float hreg[2][8];
#pragma unroll
  for (int c = 0; c < 2; ++c)
#pragma unroll
    for (int r = 0; r < 8; ++r) { creg[c][r] = 0.0f; hreg[c][r] = 0.0f; }

  __syncthreads();

  for (int s = 0; s < T; ++s) {
    const int t = d.rev ? (T - 1 - s) : s;
    v8f acc[2][4] = {};   // [hidden tile col][gate i,f,g,o]

    // ---- recurrent GEMM: h_{s-1} (64x256) @ Whh^T (256x1024) ----
#pragma unroll
    for (int kt = 0; kt < 8; ++kt) {
      v16bf a = load_a_frag(hbuf + (size_t)mb * 16 * HSTR, HSTR, kt, lane);
#pragma unroll
      for (int c = 0; c < 2; ++c) {
        const int ht = 2 * qd + c;
#pragma unroll
        for (int g = 0; g < 4; ++g) {
          v16bf b = load_b_frag(d.Whh, g * 16 + ht, kt, lane);
          acc[c][g] = __builtin_amdgcn_wmma_f32_16x16x32_bf16(
              false, a, false, b, (short)0, acc[c][g], false, false);
        }
      }
    }
    // ---- input GEMM (ctx layer): x_t (64x256) @ Wih^T (256x1024) ----
    if (d.X) {
      const bf16_t* xrow = d.X + ((size_t)(mb * 16) * T + t) * 256;
#pragma unroll
      for (int kt = 0; kt < 8; ++kt) {
        v16bf a = load_a_frag(xrow, T * 256, kt, lane);
#pragma unroll
        for (int c = 0; c < 2; ++c) {
          const int ht = 2 * qd + c;
#pragma unroll
          for (int g = 0; g < 4; ++g) {
            v16bf b = load_b_frag(d.Wih, g * 16 + ht, kt, lane);
            acc[c][g] = __builtin_amdgcn_wmma_f32_16x16x32_bf16(
                false, a, false, b, (short)0, acc[c][g], false, false);
          }
        }
      }
    }

    // ---- gate nonlinearities + state update (lane-local) ----
#pragma unroll
    for (int c = 0; c < 2; ++c) {
      const int j = (2 * qd + c) * 16 + nloc;
      const float b_i = d.bias[j],       b_f = d.bias[256 + j];
      const float b_g = d.bias[512 + j], b_o = d.bias[768 + j];
      float wi0 = 0.f, wi1 = 0.f, wf0 = 0.f, wf1 = 0.f;
      float wg0 = 0.f, wg1 = 0.f, wo0 = 0.f, wo1 = 0.f;
      if (d.Xf) {   // agg layer: Din==2, do x@Wih^T in VALU
        wi0 = d.WihF[2 * j];         wi1 = d.WihF[2 * j + 1];
        wf0 = d.WihF[2 * (256 + j)]; wf1 = d.WihF[2 * (256 + j) + 1];
        wg0 = d.WihF[2 * (512 + j)]; wg1 = d.WihF[2 * (512 + j) + 1];
        wo0 = d.WihF[2 * (768 + j)]; wo1 = d.WihF[2 * (768 + j) + 1];
      }
#pragma unroll
      for (int r = 0; r < 8; ++r) {
        const int m = mb * 16 + r + 8 * hi;
        float gi = acc[c][0][r] + b_i;
        float gf = acc[c][1][r] + b_f;
        float gg = acc[c][2][r] + b_g;
        float go = acc[c][3][r] + b_o;
        if (d.Xf) {
          const float x0 = d.Xf[((size_t)m * T + t) * 2 + 0];
          const float x1 = d.Xf[((size_t)m * T + t) * 2 + 1];
          gi += x0 * wi0 + x1 * wi1;
          gf += x0 * wf0 + x1 * wf1;
          gg += x0 * wg0 + x1 * wg1;
          go += x0 * wo0 + x1 * wo1;
        }
        float cv = sigf(gf) * creg[c][r] + sigf(gi) * tanhf(gg);
        creg[c][r] = cv;
        hreg[c][r] = sigf(go) * tanhf(cv);
      }
    }

    __syncthreads();   // all waves done reading hbuf for this step

#pragma unroll
    for (int c = 0; c < 2; ++c) {
      const int j = (2 * qd + c) * 16 + nloc;
#pragma unroll
      for (int r = 0; r < 8; ++r) {
        const int m = mb * 16 + r + 8 * hi;
        const bf16_t hb = (bf16_t)hreg[c][r];
        hbuf[m * HSTR + j] = hb;
        if (d.hs) d.hs[((size_t)m * T + t) * 256 + j] = hb;
      }
    }
    __syncthreads();   // new h visible to everyone
  }

  if (d.hT) {
#pragma unroll
    for (int c = 0; c < 2; ++c) {
      const int j = (2 * qd + c) * 16 + nloc;
#pragma unroll
      for (int r = 0; r < 8; ++r) {
        const int m = mb * 16 + r + 8 * hi;
        d.hT[(size_t)m * 256 + j] = hreg[c][r];
      }
    }
  }
}

// ---- multi-perspective cosine matching (L=1) -------------------------------
__device__ inline float wred(float x) {
#pragma unroll
  for (int o = 16; o > 0; o >>= 1) x += __shfl_xor(x, o, 32);
  return x;
}

// one wave per (b,t): mv[b,t,0] = cos(w1*v1f[b,t], w1*of[b,Tv-1])
//                     mv[b,t,1] = cos(w2*v1b[b,t], w2*ob[b,0])
__global__ void match_kernel(const bf16_t* __restrict__ v1f,
                             const bf16_t* __restrict__ v1b,
                             const bf16_t* __restrict__ of,
                             const bf16_t* __restrict__ ob,
                             int T, int Tv,
                             const float* __restrict__ w1,
                             const float* __restrict__ w2,
                             float* __restrict__ mv, int BT) {
  const int widx = (blockIdx.x * blockDim.x + threadIdx.x) >> 5;
  const int lane = threadIdx.x & 31;
  if (widx >= BT) return;
  const int b = widx / T;
  const bf16_t* a1 = v1f + (size_t)widx * 256;
  const bf16_t* a2 = v1b + (size_t)widx * 256;
  const bf16_t* b1 = of + ((size_t)b * Tv + (Tv - 1)) * 256;
  const bf16_t* b2 = ob + ((size_t)b * Tv) * 256;

  float num1 = 0, na1 = 0, nb1 = 0, num2 = 0, na2 = 0, nb2 = 0;
  const int base = lane * 8;
#pragma unroll
  for (int e = 0; e < 8; ++e) {
    const int dd = base + e;
    const float wa = w1[dd];
    const float x = wa * (float)a1[dd];
    const float y = wa * (float)b1[dd];
    num1 += x * y; na1 += x * x; nb1 += y * y;
    const float wb = w2[dd];
    const float u = wb * (float)a2[dd];
    const float v = wb * (float)b2[dd];
    num2 += u * v; na2 += u * u; nb2 += v * v;
  }
  num1 = wred(num1); na1 = wred(na1); nb1 = wred(nb1);
  num2 = wred(num2); na2 = wred(na2); nb2 = wred(nb2);
  if (lane == 0) {
    float d1 = fmaxf(sqrtf(na1) * sqrtf(nb1), 1e-8f);
    float d2 = fmaxf(sqrtf(na2) * sqrtf(nb2), 1e-8f);
    mv[(size_t)widx * 2 + 0] = num1 / d1;
    mv[(size_t)widx * 2 + 1] = num2 / d2;
  }
}

// ---- final FC layers -------------------------------------------------------
// fc1: x = concat(pTf, pTb, hTf, hTb) (64x1024) -> tanh(x @ W1^T + b1), (64x512)
__global__ void fc1_kernel(const float* __restrict__ pTf,
                           const float* __restrict__ pTb,
                           const float* __restrict__ hTf,
                           const float* __restrict__ hTb,
                           const float* __restrict__ W1,
                           const float* __restrict__ b1,
                           float* __restrict__ out) {
  const int idx = blockIdx.x * blockDim.x + threadIdx.x;   // 64*512
  const int b = idx >> 9, o = idx & 511;
  const float* parts[4] = {pTf + (size_t)b * 256, pTb + (size_t)b * 256,
                           hTf + (size_t)b * 256, hTb + (size_t)b * 256};
  float s = b1[o];
  const float* wr = W1 + (size_t)o * 1024;
#pragma unroll
  for (int p = 0; p < 4; ++p) {
    const float* px = parts[p];
    const float* pw = wr + p * 256;
    for (int k = 0; k < 256; ++k) s += px[k] * pw[k];
  }
  out[(size_t)b * 512 + o] = tanhf(s);
}

// fc2: (64x512) @ W2^T (20x512) + b2 -> (64x20)
__global__ void fc2_kernel(const float* __restrict__ x,
                           const float* __restrict__ W2,
                           const float* __restrict__ b2,
                           float* __restrict__ out) {
  const int idx = blockIdx.x * blockDim.x + threadIdx.x;
  if (idx >= 64 * 20) return;
  const int b = idx / 20, c = idx % 20;
  float s = b2[c];
  const float* px = x + (size_t)b * 512;
  const float* pw = W2 + (size_t)c * 512;
  for (int k = 0; k < 512; ++k) s += px[k] * pw[k];
  out[(size_t)b * 20 + c] = s;
}

// ---------------------------------------------------------------------------
extern "C" void kernel_launch(void* const* d_in, const int* in_sizes, int n_in,
                              void* d_out, int out_size, void* d_ws,
                              size_t ws_size, hipStream_t stream) {
  (void)in_sizes; (void)n_in; (void)out_size; (void)ws_size;
  const int*   text   = (const int*)d_in[0];
  const int*   label  = (const int*)d_in[1];
  const float* emb    = (const float*)d_in[2];
  const float* cWihFf = (const float*)d_in[3];
  const float* cWhhFf = (const float*)d_in[4];
  const float* cBihF  = (const float*)d_in[5];
  const float* cBhhF  = (const float*)d_in[6];
  const float* cWihBf = (const float*)d_in[7];
  const float* cWhhBf = (const float*)d_in[8];
  const float* cBihB  = (const float*)d_in[9];
  const float* cBhhB  = (const float*)d_in[10];
  const float* w1     = (const float*)d_in[11];
  const float* w2     = (const float*)d_in[12];
  const float* aWihF  = (const float*)d_in[13];  // (1024,2)
  const float* aWhhFf = (const float*)d_in[14];
  const float* aBihF  = (const float*)d_in[15];
  const float* aBhhF  = (const float*)d_in[16];
  const float* aWihB  = (const float*)d_in[17];  // (1024,2)
  const float* aWhhBf = (const float*)d_in[18];
  const float* aBihB  = (const float*)d_in[19];
  const float* aBhhB  = (const float*)d_in[20];
  const float* fc1W   = (const float*)d_in[21];
  const float* fc1b   = (const float*)d_in[22];
  const float* fc2W   = (const float*)d_in[23];
  const float* fc2b   = (const float*)d_in[24];
  float* out = (float*)d_out;

  // workspace carve-out (~60 MB)
  size_t off = 0;
  auto alloc = [&](size_t bytes) {
    void* p = (char*)d_ws + off;
    off += (bytes + 255) & ~(size_t)255;
    return p;
  };
  bf16_t* Xp     = (bf16_t*)alloc((size_t)64 * 512 * 256 * 2);
  bf16_t* Xh     = (bf16_t*)alloc((size_t)64 * 64 * 256 * 2);
  bf16_t* cWih_f = (bf16_t*)alloc((size_t)1024 * 256 * 2);
  bf16_t* cWih_b = (bf16_t*)alloc((size_t)1024 * 256 * 2);
  bf16_t* cWhh_f = (bf16_t*)alloc((size_t)1024 * 256 * 2);
  bf16_t* cWhh_b = (bf16_t*)alloc((size_t)1024 * 256 * 2);
  bf16_t* aWhh_f = (bf16_t*)alloc((size_t)1024 * 256 * 2);
  bf16_t* aWhh_b = (bf16_t*)alloc((size_t)1024 * 256 * 2);
  float*  bias_cf = (float*)alloc(1024 * 4);
  float*  bias_cb = (float*)alloc(1024 * 4);
  float*  bias_af = (float*)alloc(1024 * 4);
  float*  bias_ab = (float*)alloc(1024 * 4);
  bf16_t* cpf = (bf16_t*)alloc((size_t)64 * 512 * 256 * 2);
  bf16_t* cpb = (bf16_t*)alloc((size_t)64 * 512 * 256 * 2);
  bf16_t* chf = (bf16_t*)alloc((size_t)64 * 64 * 256 * 2);
  bf16_t* chb = (bf16_t*)alloc((size_t)64 * 64 * 256 * 2);
  float*  mvp = (float*)alloc((size_t)64 * 512 * 2 * 4);
  float*  mvh = (float*)alloc((size_t)64 * 64 * 2 * 4);
  float*  pTf = (float*)alloc((size_t)64 * 256 * 4);
  float*  pTb = (float*)alloc((size_t)64 * 256 * 4);
  float*  hTf = (float*)alloc((size_t)64 * 256 * 4);
  float*  hTb = (float*)alloc((size_t)64 * 256 * 4);
  float*  fc1o = (float*)alloc((size_t)64 * 512 * 4);

  // weight / bias prep
  int n = 1024 * 256;
  k_cvt_bf16<<<(n + 255) / 256, 256, 0, stream>>>(cWihFf, cWih_f, n);
  k_cvt_bf16<<<(n + 255) / 256, 256, 0, stream>>>(cWihBf, cWih_b, n);
  k_cvt_bf16<<<(n + 255) / 256, 256, 0, stream>>>(cWhhFf, cWhh_f, n);
  k_cvt_bf16<<<(n + 255) / 256, 256, 0, stream>>>(cWhhBf, cWhh_b, n);
  k_cvt_bf16<<<(n + 255) / 256, 256, 0, stream>>>(aWhhFf, aWhh_f, n);
  k_cvt_bf16<<<(n + 255) / 256, 256, 0, stream>>>(aWhhBf, aWhh_b, n);
  k_bias_add<<<4, 256, 0, stream>>>(cBihF, cBhhF, bias_cf, 1024);
  k_bias_add<<<4, 256, 0, stream>>>(cBihB, cBhhB, bias_cb, 1024);
  k_bias_add<<<4, 256, 0, stream>>>(aBihF, aBhhF, bias_af, 1024);
  k_bias_add<<<4, 256, 0, stream>>>(aBihB, aBhhB, bias_ab, 1024);

  // embedding gather -> bf16
  n = 64 * 512 * 256;
  k_gather<<<(n + 255) / 256, 256, 0, stream>>>(text, emb, Xp, n);
  n = 64 * 64 * 256;
  k_gather<<<(n + 255) / 256, 256, 0, stream>>>(label, emb, Xh, n);

  // context BiLSTM: 4 independent scans, one workgroup each
  ScanQuad q1;
  q1.s[0] = {Xp, nullptr, cWih_f, nullptr, cWhh_f, bias_cf, cpf, nullptr, 512, 0};
  q1.s[1] = {Xp, nullptr, cWih_b, nullptr, cWhh_b, bias_cb, cpb, nullptr, 512, 1};
  q1.s[2] = {Xh, nullptr, cWih_f, nullptr, cWhh_f, bias_cf, chf, nullptr, 64, 0};
  q1.s[3] = {Xh, nullptr, cWih_b, nullptr, cWhh_b, bias_cb, chb, nullptr, 64, 1};
  lstm_scan_kernel<<<4, 1024, 0, stream>>>(q1);

  // multi-perspective matching (L=1)
  match_kernel<<<(64 * 512) / 8, 256, 0, stream>>>(cpf, cpb, chf, chb, 512, 64,
                                                   w1, w2, mvp, 64 * 512);
  match_kernel<<<(64 * 64) / 8, 256, 0, stream>>>(chf, chb, cpf, cpb, 64, 512,
                                                  w1, w2, mvh, 64 * 64);

  // aggregation BiLSTM (input dim 2): only final hidden states needed
  ScanQuad q2;
  q2.s[0] = {nullptr, mvp, nullptr, aWihF, aWhh_f, bias_af, nullptr, pTf, 512, 0};
  q2.s[1] = {nullptr, mvp, nullptr, aWihB, aWhh_b, bias_ab, nullptr, pTb, 512, 1};
  q2.s[2] = {nullptr, mvh, nullptr, aWihF, aWhh_f, bias_af, nullptr, hTf, 64, 0};
  q2.s[3] = {nullptr, mvh, nullptr, aWihB, aWhh_b, bias_ab, nullptr, hTb, 64, 1};
  lstm_scan_kernel<<<4, 1024, 0, stream>>>(q2);

  // classifier head
  fc1_kernel<<<(64 * 512) / 256, 256, 0, stream>>>(pTf, pTb, hTf, hTb, fc1W,
                                                   fc1b, fc1o);
  fc2_kernel<<<(64 * 20 + 255) / 256, 256, 0, stream>>>(fc1o, fc2W, fc2b, out);
}
